// SimpleGCN_40226663694509
// MI455X (gfx1250) — matrix operand order repaired and checked by reading, not compiled, not run on hardware
//
#include <hip/hip_runtime.h>

typedef __attribute__((ext_vector_type(16))) __bf16 v16bf;
typedef __attribute__((ext_vector_type(8)))  float  v8f;

__device__ __forceinline__ float4 ldg4(const float* p) {
    return *reinterpret_cast<const float4*>(p);
}

__device__ __forceinline__ v16bf pack16(float4 a, float4 b, float4 c, float4 d) {
    v16bf v;
    v[0]  = (__bf16)a.x; v[1]  = (__bf16)a.y; v[2]  = (__bf16)a.z; v[3]  = (__bf16)a.w;
    v[4]  = (__bf16)b.x; v[5]  = (__bf16)b.y; v[6]  = (__bf16)b.z; v[7]  = (__bf16)b.w;
    v[8]  = (__bf16)c.x; v[9]  = (__bf16)c.y; v[10] = (__bf16)c.z; v[11] = (__bf16)c.w;
    v[12] = (__bf16)d.x; v[13] = (__bf16)d.y; v[14] = (__bf16)d.z; v[15] = (__bf16)d.w;
    return v;
}

// ---------------- SpMM pieces (edge-parallel, f32 atomics at L2) -------------

__global__ void gcn_deg(const int* __restrict__ row, float* __restrict__ deg, int E) {
    int e = blockIdx.x * blockDim.x + threadIdx.x;
    if (e < E)
        __hip_atomic_fetch_add(&deg[row[e]], 1.0f, __ATOMIC_RELAXED, __HIP_MEMORY_SCOPE_AGENT);
}

__global__ void gcn_dinv(float* __restrict__ deg, int Nn) {
    int i = blockIdx.x * blockDim.x + threadIdx.x;
    if (i < Nn) deg[i] = rsqrtf(deg[i] + 1.0f);   // +1 = self loop; deg>=1 so clip is no-op
}

// y[i,:] = dinv[i]^2 * x[i,:]   (self-loop term; also initializes y, no memset needed)
__global__ void gcn_selfloop(const float* __restrict__ x, const float* __restrict__ dinv,
                             float* __restrict__ y, int Nn, int dim4) {
    int t = blockIdx.x * blockDim.x + threadIdx.x;
    if (t >= Nn * dim4) return;
    int i = t / dim4;
    int f = (t - i * dim4) * 4;
    int dim = dim4 * 4;
    float w = dinv[i]; w = w * w;
    float4 v = ldg4(x + (size_t)i * dim + f);
    float4 r = make_float4(w * v.x, w * v.y, w * v.z, w * v.w);
    *reinterpret_cast<float4*>(y + (size_t)i * dim + f) = r;
}

// y[row[e],:] += dinv[row]*dinv[col] * x[col[e],:]
__global__ void gcn_spmm_edges(const int* __restrict__ row, const int* __restrict__ col,
                               const float* __restrict__ dinv, const float* __restrict__ x,
                               float* __restrict__ y, int E, int dim4) {
    int t = blockIdx.x * blockDim.x + threadIdx.x;
    if (t >= E * dim4) return;
    int e = t / dim4;
    int f = (t - e * dim4) * 4;
    int dim = dim4 * 4;
    int r = row[e], c = col[e];
    float w = dinv[r] * dinv[c];
    float4 v = ldg4(x + (size_t)c * dim + f);
    float* yp = y + (size_t)r * dim + f;
    __hip_atomic_fetch_add(yp + 0, w * v.x, __ATOMIC_RELAXED, __HIP_MEMORY_SCOPE_AGENT);
    __hip_atomic_fetch_add(yp + 1, w * v.y, __ATOMIC_RELAXED, __HIP_MEMORY_SCOPE_AGENT);
    __hip_atomic_fetch_add(yp + 2, w * v.z, __ATOMIC_RELAXED, __HIP_MEMORY_SCOPE_AGENT);
    __hip_atomic_fetch_add(yp + 3, w * v.w, __ATOMIC_RELAXED, __HIP_MEMORY_SCOPE_AGENT);
}

// ------------- Dense GEMM: out[M,N] = act(A[M,K] @ W[N,K]^T + bias) ----------
// Compile-time K and N: fully unrolled K loop, immediate-offset stores.
// Each wave computes a 16x32 output tile (two 16x16 WMMA accumulators sharing
// one A fragment -> 2x arithmetic intensity on the big activation matrix).

template<int K, int N, bool RELU>
__global__ __launch_bounds__(256) void gcn_gemm_wmma(
    const float* __restrict__ A,     // [M][K]
    const float* __restrict__ W,     // [N][K]
    const float* __restrict__ bias,  // [N]
    float* __restrict__ out,         // [M][N]
    int M)
{
    const int wave  = threadIdx.x >> 5;
    const int lane  = threadIdx.x & 31;
    const int rm    = lane & 15;     // row-of-A / col-of-W index for this lane
    const int khalf = lane >> 4;     // lane half selects K sub-group
    const int tileM = blockIdx.x * 128 + wave * 16;
    const int tileN = blockIdx.y * 32;
    if (tileM >= M) return;          // wave-uniform guard (M%16==0), EXEC stays all-ones

    v8f acc0 = {};
    v8f acc1 = {};
    // A 16-bit layout: lane half 0 -> K {0..7, 16..23}; half 1 -> K {8..15, 24..31}
    const float* arow  = A + (size_t)(tileM + rm) * K + 8 * khalf;
    // B 16-bit layout: lane half 0 -> K {0..7, 16..23}; half 1 -> K {8..15, 24..31}
    const float* brow0 = W + (size_t)(tileN + rm) * K + 8 * khalf;
    const float* brow1 = brow0 + (size_t)16 * K;

    #pragma unroll
    for (int kb = 0; kb < K; kb += 32) {
        v16bf av = pack16(ldg4(arow + kb),      ldg4(arow + kb + 4),
                          ldg4(arow + kb + 16), ldg4(arow + kb + 20));
        v16bf bv0 = pack16(ldg4(brow0 + kb),     ldg4(brow0 + kb + 4),
                           ldg4(brow0 + kb + 16), ldg4(brow0 + kb + 20));
        v16bf bv1 = pack16(ldg4(brow1 + kb),     ldg4(brow1 + kb + 4),
                           ldg4(brow1 + kb + 16), ldg4(brow1 + kb + 20));
        acc0 = __builtin_amdgcn_wmma_f32_16x16x32_bf16(
            false, av, false, bv0, (short)0, acc0, false, false);
        acc1 = __builtin_amdgcn_wmma_f32_16x16x32_bf16(
            false, av, false, bv1, (short)0, acc1, false, false);
    }

    // C/D layout: VGPR r, lane half h -> row (r + 8h), col = rm (+16 for tile 1)
    const float bb0 = bias[tileN + rm];
    const float bb1 = bias[tileN + 16 + rm];
    float* obase = out + (size_t)(tileM + 8 * khalf) * N + tileN + rm;
    #pragma unroll
    for (int r = 0; r < 8; ++r) {
        float v0 = acc0[r] + bb0;
        float v1 = acc1[r] + bb1;
        if (RELU) { v0 = fmaxf(v0, 0.0f); v1 = fmaxf(v1, 0.0f); }
        obase[(size_t)r * N]      = v0;
        obase[(size_t)r * N + 16] = v1;
    }
}

// Generic runtime-dim fallback (one 16x16 tile per wave)
template<bool RELU>
__global__ __launch_bounds__(256) void gcn_gemm_wmma_gen(
    const float* __restrict__ A, const float* __restrict__ W,
    const float* __restrict__ bias, float* __restrict__ out,
    int M, int N, int K)
{
    const int wave  = threadIdx.x >> 5;
    const int lane  = threadIdx.x & 31;
    const int rm    = lane & 15;
    const int khalf = lane >> 4;
    const int tileM = blockIdx.x * 128 + wave * 16;
    const int tileN = blockIdx.y * 16;
    if (tileM >= M) return;

    v8f acc = {};
    const float* arow = A + (size_t)(tileM + rm) * K + 8 * khalf;
    const float* brow = W + (size_t)(tileN + rm) * K + 8 * khalf;
    for (int kb = 0; kb < K; kb += 32) {
        v16bf av = pack16(ldg4(arow + kb),      ldg4(arow + kb + 4),
                          ldg4(arow + kb + 16), ldg4(arow + kb + 20));
        v16bf bv = pack16(ldg4(brow + kb),     ldg4(brow + kb + 4),
                          ldg4(brow + kb + 16), ldg4(brow + kb + 20));
        acc = __builtin_amdgcn_wmma_f32_16x16x32_bf16(
            false, av, false, bv, (short)0, acc, false, false);
    }
    const float bb = bias[tileN + rm];
    #pragma unroll
    for (int r = 0; r < 8; ++r) {
        float v = acc[r] + bb;
        if (RELU) v = fmaxf(v, 0.0f);
        out[(size_t)(tileM + r + 8 * khalf) * N + tileN + rm] = v;
    }
}

// ---------------------------------------------------------------------------

extern "C" void kernel_launch(void* const* d_in, const int* in_sizes, int n_in,
                              void* d_out, int out_size, void* d_ws, size_t ws_size,
                              hipStream_t stream) {
    const float* x  = (const float*)d_in[0];
    const int*   ei = (const int*)  d_in[1];
    const float* W1 = (const float*)d_in[2];
    const float* b1 = (const float*)d_in[3];
    const float* W2 = (const float*)d_in[4];
    const float* b2 = (const float*)d_in[5];

    const int hidden = in_sizes[3];            // 256
    const int outd   = in_sizes[5];            // 128
    const int ind    = in_sizes[2] / hidden;   // 128
    const int Nn     = in_sizes[0] / ind;      // 50000
    const int E      = in_sizes[1] / 2;        // 800000
    const int* row = ei;
    const int* col = ei + E;

    // Workspace layout (y1 and y2 alias: y1 is dead once h1 is written)
    char* p = (char*)d_ws;
    auto alloc = [&](size_t bytes) { void* q = p; p += (bytes + 255) & ~(size_t)255; return q; };
    float* dinv = (float*)alloc((size_t)Nn * 4);
    float* h1   = (float*)alloc((size_t)Nn * hidden * 4);
    float* ybuf = (float*)alloc((size_t)Nn * hidden * 4);
    float* y1 = ybuf;   // [Nn][ind]
    float* y2 = ybuf;   // [Nn][hidden]
    float* outp = (float*)d_out;

    hipMemsetAsync(dinv, 0, (size_t)Nn * 4, stream);
    gcn_deg <<<(E + 255) / 256, 256, 0, stream>>>(row, dinv, E);
    gcn_dinv<<<(Nn + 255) / 256, 256, 0, stream>>>(dinv, Nn);

    // layer 1: y1 = A @ x ; h1 = relu(y1 @ W1^T + b1)
    const int d4a = ind / 4;
    gcn_selfloop  <<<(Nn * d4a + 255) / 256, 256, 0, stream>>>(x, dinv, y1, Nn, d4a);
    gcn_spmm_edges<<<(E  * d4a + 255) / 256, 256, 0, stream>>>(row, col, dinv, x, y1, E, d4a);

    const int mblocks = (Nn + 127) / 128;
    if (ind == 128 && hidden == 256) {
        dim3 g1(mblocks, 256 / 32);
        gcn_gemm_wmma<128, 256, true><<<g1, 256, 0, stream>>>(y1, W1, b1, h1, Nn);
    } else {
        dim3 g1(mblocks, hidden / 16);
        gcn_gemm_wmma_gen<true><<<g1, 256, 0, stream>>>(y1, W1, b1, h1, Nn, hidden, ind);
    }

    // layer 2: y2 = A @ h1 ; out = y2 @ W2^T + b2
    const int d4b = hidden / 4;
    gcn_selfloop  <<<(Nn * d4b + 255) / 256, 256, 0, stream>>>(h1, dinv, y2, Nn, d4b);
    gcn_spmm_edges<<<(E  * d4b + 255) / 256, 256, 0, stream>>>(row, col, dinv, h1, y2, E, d4b);

    if (hidden == 256 && outd == 128) {
        dim3 g2(mblocks, 128 / 32);
        gcn_gemm_wmma<256, 128, false><<<g2, 256, 0, stream>>>(y2, W2, b2, outp, Nn);
    } else {
        dim3 g2(mblocks, outd / 16);
        gcn_gemm_wmma_gen<false><<<g2, 256, 0, stream>>>(y2, W2, b2, outp, Nn, outd, hidden);
    }
}
